// CoAttention_65747359367289
// MI455X (gfx1250) — matrix operand order repaired and checked by reading, not compiled, not run on hardware
//
#include <hip/hip_runtime.h>

// ---------------------------------------------------------------------------
// CDNA5 (gfx1250) co-attention pipeline.
// All GEMMs: v_wmma_f32_16x16x32_bf16, LDS double-buffered via
// global_load_async_to_lds_b128 (ASYNCcnt), all-TRANSB data layout so every
// staging transfer is a contiguous 16B async copy.
// ---------------------------------------------------------------------------

typedef __bf16 bf16;
typedef __attribute__((ext_vector_type(16))) __bf16 v16bf;
typedef __attribute__((ext_vector_type(8)))  __bf16 bf16x8;
typedef __attribute__((ext_vector_type(8)))  float  v8f;

#define BM 128
#define BN 128
#define BK 32
#define KPAD 8                 // row stride 40 bf16 = 80 B (16B-aligned rows)
#define LDST (BK + KPAD)

static __device__ __forceinline__ bf16 f2bf(float x) {
  unsigned u; __builtin_memcpy(&u, &x, 4);
  unsigned r = u + 0x7FFFu + ((u >> 16) & 1u);   // round-to-nearest-even
  unsigned short h = (unsigned short)(r >> 16);
  bf16 o; __builtin_memcpy(&o, &h, 2); return o;
}
static __device__ __forceinline__ float bf2f(bf16 h) {
  unsigned short u; __builtin_memcpy(&u, &h, 2);
  unsigned x = ((unsigned)u) << 16;
  float f; __builtin_memcpy(&f, &x, 4); return f;
}
static __device__ __forceinline__ void storeOut(bf16* p, float v)  { *p = f2bf(v); }
static __device__ __forceinline__ void storeOut(float* p, float v) { *p = v; }

// --- CDNA5 async copy: 16 B per lane, global -> LDS, tracked by ASYNCcnt ---
static __device__ __forceinline__ void async_copy16(unsigned lds_addr,
                                                    const bf16* gsrc) {
  asm volatile("global_load_async_to_lds_b128 %0, %1, off"
               :: "v"(lds_addr), "v"((unsigned long long)(size_t)gsrc)
               : "memory");
}
static __device__ __forceinline__ void wait_async0() {
  asm volatile("s_wait_asynccnt 0" ::: "memory");
}

// ---------------------------------------------------------------------------
// Tiled batched GEMM (always B^T layout):
//   Out[b] = op(A[b] @ B[b]^T (+bias)),  op in {id, tanh}
//   A:  M x K row-major bf16
//   Bm: N x K row-major bf16 (K-contiguous -> contiguous async staging)
//   Out: M x N (OT = bf16 or float)
// 256 threads = 8 waves; block tile 128x128, K step 32, 2-deep LDS pipeline.
// Requires M%128==0, N%128==0, K%32==0 (true for every stage here).
// ---------------------------------------------------------------------------
template <typename OT, bool BIAS, bool TANH>
__global__ __launch_bounds__(256) void gemm_wmma(
    const bf16* __restrict__ A, size_t strideA, int lda,
    const bf16* __restrict__ Bm, size_t strideB, int ldb,
    const float* __restrict__ bias,
    OT* __restrict__ Out, size_t strideO, int ldo,
    int M, int N, int K)
{
  __shared__ __align__(16) bf16 As[2][BM][LDST];
  __shared__ __align__(16) bf16 Bs[2][BN][LDST];

  const int tid  = threadIdx.x;
  const int lane = tid & 31;
  const int wave = tid >> 5;
  const int wm   = wave >> 2;          // 0..1
  const int wn   = wave & 3;           // 0..3
  const int kg   = lane >> 4;          // K half selector
  const int l15  = lane & 15;

  const int bz   = blockIdx.z;
  const int row0 = blockIdx.y * BM;
  const int col0 = blockIdx.x * BN;

  const bf16* Ab = A  + (size_t)bz * strideA;
  const bf16* Bb = Bm + (size_t)bz * strideB;
  OT*         Ob = Out + (size_t)bz * strideO;

  v8f acc[4][2];
#pragma unroll
  for (int i = 0; i < 4; ++i)
#pragma unroll
    for (int j = 0; j < 2; ++j)
#pragma unroll
      for (int r = 0; r < 8; ++r) acc[i][j][r] = 0.0f;

  // staging: 128 rows x 64 B per tile; each thread owns 32 B of one row
  const int sr = tid >> 1;             // 0..127
  const int sc = (tid & 1) * 16;       // 0 or 16 (bf16 elems)
  const unsigned asBase = (unsigned)(size_t)&As[0][0][0];
  const unsigned bsBase = (unsigned)(size_t)&Bs[0][0][0];

  const int nK = K / BK;

  auto stage = [&](int q, int k0) {
    const bf16* ga = Ab + (size_t)(row0 + sr) * lda + (k0 + sc);
    unsigned la = asBase + (unsigned)(((q * BM + sr) * LDST + sc) * 2);
    async_copy16(la,      ga);
    async_copy16(la + 16, ga + 8);
    const bf16* gb = Bb + (size_t)(col0 + sr) * ldb + (k0 + sc);
    unsigned lb = bsBase + (unsigned)(((q * BN + sr) * LDST + sc) * 2);
    async_copy16(lb,      gb);
    async_copy16(lb + 16, gb + 8);
  };

  stage(0, 0);

  for (int kt = 0; kt < nK; ++kt) {
    wait_async0();          // this wave's copies for buffer (kt&1) landed
    __syncthreads();        // everyone's copies landed
    if (kt + 1 < nK) stage((kt + 1) & 1, (kt + 1) * BK);  // overlap next tile

    const int q = kt & 1;

    // ---- fragments per documented CDNA5 16-bit layouts ----
    // A 16x32: lane group kg holds K in {kg*8..+7} and {16+kg*8..+7}
    v16bf afrag[4], bfrag[2];
#pragma unroll
    for (int tr = 0; tr < 4; ++tr) {
      const int r = wm * 64 + tr * 16 + l15;
      bf16x8 lo = *(const bf16x8*)&As[q][r][kg * 8];
      bf16x8 hi = *(const bf16x8*)&As[q][r][16 + kg * 8];
#pragma unroll
      for (int i = 0; i < 8; ++i) { afrag[tr][i] = lo[i]; afrag[tr][8 + i] = hi[i]; }
    }
    // B 32x16: col = lane&15, lane group kg holds K in {kg*8..+7} and {16+kg*8..+7}
#pragma unroll
    for (int tc = 0; tc < 2; ++tc) {
      const int c = wn * 32 + tc * 16 + l15;
      bf16x8 lo = *(const bf16x8*)&Bs[q][c][kg * 8];
      bf16x8 hi = *(const bf16x8*)&Bs[q][c][16 + kg * 8];
#pragma unroll
      for (int i = 0; i < 8; ++i) { bfrag[tc][i] = lo[i]; bfrag[tc][8 + i] = hi[i]; }
    }

#pragma unroll
    for (int tr = 0; tr < 4; ++tr)
#pragma unroll
      for (int tc = 0; tc < 2; ++tc)
        acc[tr][tc] = __builtin_amdgcn_wmma_f32_16x16x32_bf16(
            false, afrag[tr], false, bfrag[tc], (short)0, acc[tr][tc],
            false, false);

    __syncthreads();        // buffer q fully consumed before it is refilled
  }

  // ---- epilogue: C/D layout -> (m = r + 8*kg, n = lane&15) ----
#pragma unroll
  for (int tr = 0; tr < 4; ++tr) {
#pragma unroll
    for (int tc = 0; tc < 2; ++tc) {
      const int gcol = col0 + wn * 32 + tc * 16 + l15;
      float bv = BIAS ? bias[gcol] : 0.0f;
#pragma unroll
      for (int r = 0; r < 8; ++r) {
        const int grow = row0 + wm * 64 + tr * 16 + r + kg * 8;
        float v = acc[tr][tc][r];
        if (BIAS) v += bv;
        if (TANH) v = tanhf(v);
        storeOut(&Ob[(size_t)grow * ldo + gcol], v);
      }
    }
  }
}

// ---------------------------------------------------------------------------
// fp32 -> bf16 (plain)
// ---------------------------------------------------------------------------
__global__ __launch_bounds__(256) void f32_to_bf16(
    const float* __restrict__ in, bf16* __restrict__ out, size_t n)
{
  size_t i = (size_t)blockIdx.x * blockDim.x + threadIdx.x;
  if (i < n) out[i] = f2bf(in[i]);
}

// ---------------------------------------------------------------------------
// fp32 K x N -> bf16 N x K (transpose-convert, for weight matrices)
// ---------------------------------------------------------------------------
__global__ __launch_bounds__(256) void f32_to_bf16_T(
    const float* __restrict__ in, bf16* __restrict__ out, int K, int N)
{
  size_t j = (size_t)blockIdx.x * blockDim.x + threadIdx.x;
  if (j >= (size_t)K * N) return;
  int n = (int)(j / K);
  int k = (int)(j % K);
  out[j] = f2bf(in[(size_t)k * N + n]);
}

// ---------------------------------------------------------------------------
// Batched bf16 transpose: in (B, R, C) -> out (B, C, R); 32x32 LDS tiles.
// grid (C/32, R/32, B), block (32, 8)
// ---------------------------------------------------------------------------
__global__ __launch_bounds__(256) void transpose_bf16(
    const bf16* __restrict__ in, bf16* __restrict__ out, int R, int C)
{
  __shared__ bf16 t[32][33];
  const int b  = blockIdx.z;
  const int tx = threadIdx.x, ty = threadIdx.y;
  const bf16* ib = in  + (size_t)b * R * C;
  bf16*       ob = out + (size_t)b * C * R;
  const int x  = blockIdx.x * 32 + tx;     // column in input
  const int y0 = blockIdx.y * 32;          // row base in input
#pragma unroll
  for (int j = 0; j < 4; ++j)
    t[ty + 8 * j][tx] = ib[(size_t)(y0 + ty + 8 * j) * C + x];
  __syncthreads();
  const int xo = blockIdx.y * 32 + tx;     // column in output (= input row)
  const int yo = blockIdx.x * 32;          // row base in output (= input col)
#pragma unroll
  for (int j = 0; j < 4; ++j)
    ob[(size_t)(yo + ty + 8 * j) * R + xo] = t[tx][ty + 8 * j];
}

// ---------------------------------------------------------------------------
// Row softmax over bf16 rows of length L (L == 2048 here, 8 elems/thread)
// ---------------------------------------------------------------------------
__global__ __launch_bounds__(256) void softmax_rows(
    const bf16* __restrict__ C, bf16* __restrict__ Out, int L)
{
  __shared__ float red[8];
  const size_t base = (size_t)blockIdx.x * L;
  const int tid = threadIdx.x;
  const int per = L / 256;
  float vals[8];
  float m = -3.4e38f;
  for (int i = 0; i < per; ++i) {
    float v = bf2f(C[base + tid + i * 256]);
    vals[i] = v; m = fmaxf(m, v);
  }
  for (int o = 16; o; o >>= 1) m = fmaxf(m, __shfl_xor(m, o, 32));
  if ((tid & 31) == 0) red[tid >> 5] = m;
  __syncthreads();
  m = red[0];
  for (int i = 1; i < 8; ++i) m = fmaxf(m, red[i]);
  __syncthreads();
  float s = 0.f;
  for (int i = 0; i < per; ++i) { vals[i] = __expf(vals[i] - m); s += vals[i]; }
  for (int o = 16; o; o >>= 1) s += __shfl_xor(s, o, 32);
  if ((tid & 31) == 0) red[tid >> 5] = s;
  __syncthreads();
  s = 0.f;
  for (int i = 0; i < 8; ++i) s += red[i];
  const float inv = 1.0f / s;
  for (int i = 0; i < per; ++i) Out[base + tid + i * 256] = f2bf(vals[i] * inv);
}

// ---------------------------------------------------------------------------
// out = LayerNorm(z + zco) * g + beta ; one block per row of length D
// ---------------------------------------------------------------------------
__global__ __launch_bounds__(256) void residual_ln(
    const float* __restrict__ z, const float* __restrict__ zco,
    const float* __restrict__ g, const float* __restrict__ be,
    float* __restrict__ out, int D)
{
  __shared__ float rs[8];
  __shared__ float rss[8];
  const size_t base = (size_t)blockIdx.x * D;
  const int tid = threadIdx.x;
  float s = 0.f, ss = 0.f;
  for (int i = tid; i < D; i += 256) {
    float v = z[base + i] + zco[base + i];
    s += v; ss += v * v;
  }
  for (int o = 16; o; o >>= 1) { s += __shfl_xor(s, o, 32); ss += __shfl_xor(ss, o, 32); }
  if ((tid & 31) == 0) { rs[tid >> 5] = s; rss[tid >> 5] = ss; }
  __syncthreads();
  s = 0.f; ss = 0.f;
  for (int i = 0; i < 8; ++i) { s += rs[i]; ss += rss[i]; }
  const float mu  = s / (float)D;
  const float var = ss / (float)D - mu * mu;
  const float inv = rsqrtf(var + 1e-5f);
  for (int i = tid; i < D; i += 256) {
    float v = z[base + i] + zco[base + i];
    out[base + i] = (v - mu) * inv * g[i] + be[i];
  }
}

// ---------------------------------------------------------------------------
// Host-side orchestration
// ---------------------------------------------------------------------------
extern "C" void kernel_launch(void* const* d_in, const int* in_sizes, int n_in,
                              void* d_out, int out_size, void* d_ws, size_t ws_size,
                              hipStream_t stream)
{
  (void)in_sizes; (void)n_in; (void)out_size; (void)ws_size;

  const float* z_a = (const float*)d_in[0];
  const float* z_b = (const float*)d_in[1];
  const float* Wa  = (const float*)d_in[2];
  const float* ba  = (const float*)d_in[3];
  const float* Wb  = (const float*)d_in[4];
  const float* bb  = (const float*)d_in[5];
  const float* Wco = (const float*)d_in[6];
  const float* Woa = (const float*)d_in[7];
  const float* boa = (const float*)d_in[8];
  const float* Wob = (const float*)d_in[9];
  const float* bob = (const float*)d_in[10];
  const float* ga  = (const float*)d_in[11];
  const float* bea = (const float*)d_in[12];
  const float* gb  = (const float*)d_in[13];
  const float* beb = (const float*)d_in[14];

  constexpr int NB = 16, LA = 2048, LB = 2048, DA = 768, DB = 512, DH = 512;

  char* wsp = (char*)d_ws;
  auto alloc = [&](size_t bytes) -> char* {
    char* p = wsp; wsp += (bytes + 255) & ~(size_t)255; return p;
  };
  const size_t sWa = (size_t)DA * DH, sWb = (size_t)DB * DH, sWc = (size_t)DH * DH;
  const size_t sWoa = (size_t)DH * DA, sWob = (size_t)DH * DB;

  bf16* z_a_bf = (bf16*)alloc((size_t)NB * LA * DA * 2);
  bf16* z_b_bf = (bf16*)alloc((size_t)NB * LB * DB * 2);
  bf16* WaT  = (bf16*)alloc(sWa * 2);    // DH x DA
  bf16* WbT  = (bf16*)alloc(sWb * 2);    // DH x DB
  bf16* WcoT = (bf16*)alloc(sWc * 2);    // DH x DH
  bf16* WoaT = (bf16*)alloc(sWoa * 2);   // DA x DH
  bf16* WobT = (bf16*)alloc(sWob * 2);   // DB x DH
  bf16* h_a  = (bf16*)alloc((size_t)NB * LA * DH * 2);
  bf16* h_b  = (bf16*)alloc((size_t)NB * LB * DH * 2);
  bf16* h_aT = (bf16*)alloc((size_t)NB * DH * LA * 2);
  bf16* h_bT = (bf16*)alloc((size_t)NB * DH * LB * 2);
  bf16* hW   = (bf16*)alloc((size_t)NB * LA * DH * 2);
  bf16* Cm   = (bf16*)alloc((size_t)NB * LA * LB * 2);
  bf16* CmT  = (bf16*)alloc((size_t)NB * LB * LA * 2);
  bf16* atA  = (bf16*)alloc((size_t)NB * LA * LB * 2);
  bf16* atB  = (bf16*)alloc((size_t)NB * LB * LA * 2);
  bf16* haco = (bf16*)alloc((size_t)NB * LA * DH * 2);
  bf16* hbco = (bf16*)alloc((size_t)NB * LB * DH * 2);
  float* zaco = (float*)alloc((size_t)NB * LA * DA * 4);
  float* zbco = (float*)alloc((size_t)NB * LB * DB * 4);

  float* out_a = (float*)d_out;
  float* out_b = out_a + (size_t)NB * LA * DA;

  const dim3 blk(256);
  const dim3 tblk(32, 8);

  // activations -> bf16 ; weights -> transposed bf16 ([N][K])
  const size_t nza = (size_t)NB * LA * DA, nzb = (size_t)NB * LB * DB;
  f32_to_bf16<<<(unsigned)((nza + 255) / 256), blk, 0, stream>>>(z_a, z_a_bf, nza);
  f32_to_bf16<<<(unsigned)((nzb + 255) / 256), blk, 0, stream>>>(z_b, z_b_bf, nzb);
  f32_to_bf16_T<<<(unsigned)((sWa  + 255) / 256), blk, 0, stream>>>(Wa,  WaT,  DA, DH);
  f32_to_bf16_T<<<(unsigned)((sWb  + 255) / 256), blk, 0, stream>>>(Wb,  WbT,  DB, DH);
  f32_to_bf16_T<<<(unsigned)((sWc  + 255) / 256), blk, 0, stream>>>(Wco, WcoT, DH, DH);
  f32_to_bf16_T<<<(unsigned)((sWoa + 255) / 256), blk, 0, stream>>>(Woa, WoaT, DH, DA);
  f32_to_bf16_T<<<(unsigned)((sWob + 255) / 256), blk, 0, stream>>>(Wob, WobT, DH, DB);

  // h_a = z_a @ Wa + ba ; h_b = z_b @ Wb + bb
  gemm_wmma<bf16, true, false><<<dim3(DH / BN, LA / BM, NB), blk, 0, stream>>>(
      z_a_bf, (size_t)LA * DA, DA, WaT, 0, DA, ba,
      h_a, (size_t)LA * DH, DH, LA, DH, DA);
  gemm_wmma<bf16, true, false><<<dim3(DH / BN, LB / BM, NB), blk, 0, stream>>>(
      z_b_bf, (size_t)LB * DB, DB, WbT, 0, DB, bb,
      h_b, (size_t)LB * DH, DH, LB, DH, DB);

  // transposed activations for later B^T use
  transpose_bf16<<<dim3(DH / 32, LA / 32, NB), tblk, 0, stream>>>(h_a, h_aT, LA, DH);
  transpose_bf16<<<dim3(DH / 32, LB / 32, NB), tblk, 0, stream>>>(h_b, h_bT, LB, DH);

  // hW = h_a @ W_co
  gemm_wmma<bf16, false, false><<<dim3(DH / BN, LA / BM, NB), blk, 0, stream>>>(
      h_a, (size_t)LA * DH, DH, WcoT, 0, DH, nullptr,
      hW, (size_t)LA * DH, DH, LA, DH, DH);

  // C = tanh(hW @ h_b^T)   (h_b already [N=LB][K=DH])
  gemm_wmma<bf16, false, true><<<dim3(LB / BN, LA / BM, NB), blk, 0, stream>>>(
      hW, (size_t)LA * DH, DH, h_b, (size_t)LB * DH, DH, nullptr,
      Cm, (size_t)LA * LB, LB, LA, LB, DH);

  // attn_a = rowsoftmax(C) ; attn_b = rowsoftmax(C^T) via tiled transpose
  softmax_rows<<<NB * LA, blk, 0, stream>>>(Cm, atA, LB);
  transpose_bf16<<<dim3(LB / 32, LA / 32, NB), tblk, 0, stream>>>(Cm, CmT, LA, LB);
  softmax_rows<<<NB * LB, blk, 0, stream>>>(CmT, atB, LA);

  // h_a_co = attn_a @ h_b  (B^T form: h_bT is [DH][LB])
  gemm_wmma<bf16, false, false><<<dim3(DH / BN, LA / BM, NB), blk, 0, stream>>>(
      atA, (size_t)LA * LB, LB, h_bT, (size_t)DH * LB, LB, nullptr,
      haco, (size_t)LA * DH, DH, LA, DH, LB);

  // h_b_co = attn_b @ h_a  (B^T form: h_aT is [DH][LA])
  gemm_wmma<bf16, false, false><<<dim3(DH / BN, LB / BM, NB), blk, 0, stream>>>(
      atB, (size_t)LB * LA, LA, h_aT, (size_t)DH * LA, LA, nullptr,
      hbco, (size_t)LB * DH, DH, LB, DH, LA);

  // z_a_co = h_a_co @ Woa + boa (fp32) ; z_b_co = h_b_co @ Wob + bob (fp32)
  gemm_wmma<float, true, false><<<dim3(DA / BN, LA / BM, NB), blk, 0, stream>>>(
      haco, (size_t)LA * DH, DH, WoaT, 0, DH, boa,
      zaco, (size_t)LA * DA, DA, LA, DA, DH);
  gemm_wmma<float, true, false><<<dim3(DB / BN, LB / BM, NB), blk, 0, stream>>>(
      hbco, (size_t)LB * DH, DH, WobT, 0, DH, bob,
      zbco, (size_t)LB * DB, DB, LB, DB, DH);

  // residual + LayerNorm
  residual_ln<<<NB * LA, blk, 0, stream>>>(z_a, zaco, ga, bea, out_a, DA);
  residual_ln<<<NB * LB, blk, 0, stream>>>(z_b, zbco, gb, beb, out_b, DB);
}
